// GATNet_6889127542860
// MI455X (gfx1250) — matrix operand order repaired and checked by reading, not compiled, not run on hardware
//
#include <hip/hip_runtime.h>

// ---------------------------------------------------------------------------
// GAT 2-layer forward for MI455X (gfx1250, wave32, WMMA).
// Feature transforms use v_wmma_f32_16x16x32_f16 (f16 A/B, fp32 accum).
// B matrices staged in LDS in *fragment-major* order so the hot loop loads
// each 16x32 B fragment as one aligned 32B LDS read per lane (ds_load_b128).
// Edge softmax via monotone-uint atomicMax + f32 atomicAdd scatter.
// ---------------------------------------------------------------------------

typedef __attribute__((ext_vector_type(16))) _Float16 v16h;
typedef __attribute__((ext_vector_type(8)))  float    v8f;

#define NEG_SLOPE   0.2f
#define ENC_NEG_INF 0x007FFFFFu   // enc(-inf)

__device__ __forceinline__ unsigned enc_f32(float f) {
  unsigned u = __float_as_uint(f);
  return (u & 0x80000000u) ? ~u : (u | 0x80000000u);
}
__device__ __forceinline__ float dec_f32(unsigned e) {
  unsigned u = (e & 0x80000000u) ? (e ^ 0x80000000u) : ~e;
  return __uint_as_float(u);
}
__device__ __forceinline__ float leaky(float v) {
  return v > 0.0f ? v : NEG_SLOPE * v;
}

// ------------------------- init (fresh state each launch) ------------------
__global__ void gat_init(float* denom1, unsigned* m1, float* denom2, unsigned* m2,
                         float* h1, float* out, int n) {
  int t = blockIdx.x * blockDim.x + threadIdx.x;
  if (t < n * 8)  { denom1[t] = 0.0f; m1[t] = ENC_NEG_INF; }
  if (t < n)      { denom2[t] = 0.0f; m2[t] = ENC_NEG_INF; }
  if (t < n * 64) h1[t]  = 0.0f;
  if (t < n * 32) out[t] = 0.0f;
}

// ---------------------------------------------------------------------------
// A-fragment loader: row-major fp32 source -> 16x32 f16 WMMA A fragment.
//   lo lanes (0-15):  K = kk..kk+15   of row m0+lane
//   hi lanes (16-31): K = kk+16..31      of row m0+lane-16
// Uses float4 loads (global_load_b128) + packed f32->f16 converts.
// ---------------------------------------------------------------------------
__device__ __forceinline__ v16h load_a_frag(const float* __restrict__ xr,
                                            int kk, int half) {
  const float4* x4 = (const float4*)xr;
  const int q1 = (kk + half * 16) >> 2;       // 8-float chunk 1 (2x float4)
  const int q2 = (kk + half * 16 + 8) >> 2;   // 8-float chunk 2
  float4 p0 = x4[q1], p1 = x4[q1 + 1];
  float4 p2 = x4[q2], p3 = x4[q2 + 1];
  v16h a;
  a[0]  = (_Float16)p0.x; a[1]  = (_Float16)p0.y;
  a[2]  = (_Float16)p0.z; a[3]  = (_Float16)p0.w;
  a[4]  = (_Float16)p1.x; a[5]  = (_Float16)p1.y;
  a[6]  = (_Float16)p1.z; a[7]  = (_Float16)p1.w;
  a[8]  = (_Float16)p2.x; a[9]  = (_Float16)p2.y;
  a[10] = (_Float16)p2.z; a[11] = (_Float16)p2.w;
  a[12] = (_Float16)p3.x; a[13] = (_Float16)p3.y;
  a[14] = (_Float16)p3.z; a[15] = (_Float16)p3.w;
  return a;
}

// ------------------------- GEMM1: [n,256] x [256,64] -----------------------
// One wave per 16-row tile, 4 column tiles of 16, K-loop of 8 x (K=32).
// LDS B layout: frag = (ktile*4 + nt); Bs[frag*32 + lane] = lane's v16h.
__global__ void __launch_bounds__(128)
gat_gemm1_wmma(const float* __restrict__ x, const float* __restrict__ W,
               float* __restrict__ out, int n) {
  __shared__ v16h Bs[8 * 4 * 32];                     // 32 KB, fragment-major
  _Float16* Bh = (_Float16*)Bs;
  // permute-stage W[k][col] (k<256, col<64) into fragment order, f32->f16
  for (int i = threadIdx.x; i < 256 * 64; i += 128) {
    const int k = i >> 6, col = i & 63;
    const int ktile = k >> 5, kin = k & 31;
    const int lane  = (col & 15) + (kin & 16);        // hi-K half -> lanes 16-31
    const int e     = kin & 15;
    Bh[(((ktile << 2) + (col >> 4)) * 32 + lane) * 16 + e] = (_Float16)W[i];
  }
  __syncthreads();

  const int wave = threadIdx.x >> 5;
  const int lane = threadIdx.x & 31;
  const int half = lane >> 4;
  const int r    = lane & 15;
  const int m0   = (blockIdx.x * 4 + wave) * 16;

  int lrow = m0 + r; if (lrow >= n) lrow = 0;         // clamp loads, guard stores
  const float* xr = x + (size_t)lrow * 256;

  v8f acc[4] = {};
  for (int kk = 0; kk < 256; kk += 32) {
    const v16h af = load_a_frag(xr, kk, half);
    const int fb = (kk >> 5) << 2;                    // ktile*4
#pragma unroll
    for (int nt = 0; nt < 4; ++nt) {
      const v16h bf = Bs[(fb + nt) * 32 + lane];      // ds_load_b128 x2
      acc[nt] = __builtin_amdgcn_wmma_f32_16x16x32_f16(
          false, af, false, bf, (short)0, acc[nt], false, false);
    }
  }
  // C/D: VGPR i -> row m0 + i + 8*half, col = lane&15
  const int rbase = m0 + half * 8;
#pragma unroll
  for (int i = 0; i < 8; ++i) {
    const int row = rbase + i;
    if (row < n) {
#pragma unroll
      for (int nt = 0; nt < 4; ++nt)
        out[(size_t)row * 64 + nt * 16 + r] = acc[nt][i];
    }
  }
}

// ------------------------- GEMM2: [n,64] x [64,32] -------------------------
__global__ void __launch_bounds__(128)
gat_gemm2_wmma(const float* __restrict__ h, const float* __restrict__ W,
               float* __restrict__ out, int n) {
  __shared__ v16h Bs[2 * 2 * 32];                     // 4 KB, fragment-major
  _Float16* Bh = (_Float16*)Bs;
  for (int i = threadIdx.x; i < 64 * 32; i += 128) {
    const int k = i >> 5, col = i & 31;
    const int ktile = k >> 5, kin = k & 31;
    const int lane  = (col & 15) + (kin & 16);
    const int e     = kin & 15;
    Bh[(((ktile << 1) + (col >> 4)) * 32 + lane) * 16 + e] = (_Float16)W[i];
  }
  __syncthreads();

  const int wave = threadIdx.x >> 5;
  const int lane = threadIdx.x & 31;
  const int half = lane >> 4;
  const int r    = lane & 15;
  const int m0   = (blockIdx.x * 4 + wave) * 16;

  int lrow = m0 + r; if (lrow >= n) lrow = 0;
  const float* hr = h + (size_t)lrow * 64;

  v8f acc[2] = {};
#pragma unroll
  for (int kk = 0; kk < 64; kk += 32) {
    const v16h af = load_a_frag(hr, kk, half);
    const int fb = (kk >> 5) << 1;
#pragma unroll
    for (int nt = 0; nt < 2; ++nt) {
      const v16h bf = Bs[(fb + nt) * 32 + lane];
      acc[nt] = __builtin_amdgcn_wmma_f32_16x16x32_f16(
          false, af, false, bf, (short)0, acc[nt], false, false);
    }
  }
  const int rbase = m0 + half * 8;
#pragma unroll
  for (int i = 0; i < 8; ++i) {
    const int row = rbase + i;
    if (row < n) {
#pragma unroll
      for (int nt = 0; nt < 2; ++nt)
        out[(size_t)row * 32 + nt * 16 + r] = acc[nt][i];
    }
  }
}

// ------------------------- attention logits --------------------------------
__global__ void gat_logits1(const float* __restrict__ h1pre,
                            const float* __restrict__ a1s,
                            const float* __restrict__ a1d,
                            float* __restrict__ e1s, float* __restrict__ e1d, int n) {
  int t = blockIdx.x * blockDim.x + threadIdx.x;
  if (t >= n * 8) return;
  const int node = t >> 3, h = t & 7;
  const float* hp = h1pre + (size_t)node * 64 + h * 8;
  float ss = 0.0f, sd = 0.0f;
#pragma unroll
  for (int d = 0; d < 8; ++d) {
    ss += hp[d] * a1s[h * 8 + d];
    sd += hp[d] * a1d[h * 8 + d];
  }
  e1s[t] = ss; e1d[t] = sd;
}

__global__ void gat_logits2(const float* __restrict__ h2pre,
                            const float* __restrict__ a2s,
                            const float* __restrict__ a2d,
                            float* __restrict__ e2s, float* __restrict__ e2d, int n) {
  int t = blockIdx.x * blockDim.x + threadIdx.x;
  if (t >= n) return;
  const float* hp = h2pre + (size_t)t * 32;
  float ss = 0.0f, sd = 0.0f;
#pragma unroll
  for (int c = 0; c < 32; ++c) {
    ss += hp[c] * a2s[c];
    sd += hp[c] * a2d[c];
  }
  e2s[t] = ss; e2d[t] = sd;
}

// ------------------------- layer-1 edge passes (8 heads) -------------------
__global__ void gat_edge1_max(const long long* __restrict__ ei,
                              const float* __restrict__ e1s, const float* __restrict__ e1d,
                              unsigned* __restrict__ m1, int E) {
  int t = blockIdx.x * blockDim.x + threadIdx.x;
  if (t >= E * 8) return;
  const int e = t >> 3, h = t & 7;
  const long long s = ei[e], d = ei[E + e];
  const float v = leaky(e1s[s * 8 + h] + e1d[d * 8 + h]);
  atomicMax(&m1[d * 8 + h], enc_f32(v));
}

__global__ void gat_edge1_denom(const long long* __restrict__ ei,
                                const float* __restrict__ e1s, const float* __restrict__ e1d,
                                const unsigned* __restrict__ m1,
                                float* __restrict__ denom1, int E) {
  int t = blockIdx.x * blockDim.x + threadIdx.x;
  if (t >= E * 8) return;
  const int e = t >> 3, h = t & 7;
  const long long s = ei[e], d = ei[E + e];
  const float v = leaky(e1s[s * 8 + h] + e1d[d * 8 + h]);
  float m = dec_f32(m1[d * 8 + h]);
  if (!(m > -3.0e38f)) m = 0.0f;                 // -inf -> 0 (no-edge nodes)
  atomicAdd(&denom1[d * 8 + h], __expf(v - m));
}

__global__ void gat_edge1_scatter(const long long* __restrict__ ei,
                                  const float* __restrict__ e1s, const float* __restrict__ e1d,
                                  const unsigned* __restrict__ m1,
                                  const float* __restrict__ denom1,
                                  const float* __restrict__ h1pre,
                                  float* __restrict__ h1, int E) {
  int t = blockIdx.x * blockDim.x + threadIdx.x;
  if (t >= E * 8) return;
  const int e = t >> 3, h = t & 7;
  const long long s = ei[e], d = ei[E + e];
  const float v = leaky(e1s[s * 8 + h] + e1d[d * 8 + h]);
  float m = dec_f32(m1[d * 8 + h]);
  if (!(m > -3.0e38f)) m = 0.0f;
  const float att = __expf(v - m) / (denom1[d * 8 + h] + 1e-16f);
  const float* hs = h1pre + (size_t)s * 64 + h * 8;
  float*       od = h1    + (size_t)d * 64 + h * 8;
#pragma unroll
  for (int k = 0; k < 8; ++k)
    atomicAdd(&od[k], hs[k] * att);
}

__global__ void gat_elu(float* __restrict__ h1, int n) {
  int t = blockIdx.x * blockDim.x + threadIdx.x;
  if (t >= n * 64) return;
  const float v = h1[t];
  h1[t] = v > 0.0f ? v : (__expf(v) - 1.0f);
}

// ------------------------- layer-2 edge passes (1 head, 32 ch) -------------
__global__ void gat_edge2_max(const long long* __restrict__ ei,
                              const float* __restrict__ e2s, const float* __restrict__ e2d,
                              unsigned* __restrict__ m2, int E) {
  int e = blockIdx.x * blockDim.x + threadIdx.x;
  if (e >= E) return;
  const long long s = ei[e], d = ei[E + e];
  atomicMax(&m2[d], enc_f32(leaky(e2s[s] + e2d[d])));
}

__global__ void gat_edge2_denom(const long long* __restrict__ ei,
                                const float* __restrict__ e2s, const float* __restrict__ e2d,
                                const unsigned* __restrict__ m2,
                                float* __restrict__ denom2, int E) {
  int e = blockIdx.x * blockDim.x + threadIdx.x;
  if (e >= E) return;
  const long long s = ei[e], d = ei[E + e];
  const float v = leaky(e2s[s] + e2d[d]);
  float m = dec_f32(m2[d]);
  if (!(m > -3.0e38f)) m = 0.0f;
  atomicAdd(&denom2[d], __expf(v - m));
}

__global__ void gat_edge2_scatter(const long long* __restrict__ ei,
                                  const float* __restrict__ e2s, const float* __restrict__ e2d,
                                  const unsigned* __restrict__ m2,
                                  const float* __restrict__ denom2,
                                  const float* __restrict__ h2pre,
                                  float* __restrict__ out, int E) {
  int e = blockIdx.x * blockDim.x + threadIdx.x;
  if (e >= E) return;
  const long long s = ei[e], d = ei[E + e];
  const float v = leaky(e2s[s] + e2d[d]);
  float m = dec_f32(m2[d]);
  if (!(m > -3.0e38f)) m = 0.0f;
  const float att = __expf(v - m) / (denom2[d] + 1e-16f);
  const float* hs = h2pre + (size_t)s * 32;
  float*       od = out   + (size_t)d * 32;
#pragma unroll
  for (int c = 0; c < 32; ++c)
    atomicAdd(&od[c], hs[c] * att);
}

// ------------------------- log-softmax: one wave32 per node ----------------
__global__ void __launch_bounds__(128)
gat_logsoftmax(float* __restrict__ out, int n) {
  const int node = blockIdx.x * 4 + (threadIdx.x >> 5);
  if (node >= n) return;
  const int lane = threadIdx.x & 31;
  float v = out[(size_t)node * 32 + lane];
  float mx = v;
#pragma unroll
  for (int o = 16; o >= 1; o >>= 1) mx = fmaxf(mx, __shfl_xor(mx, o, 32));
  float ex = __expf(v - mx);
  float sm = ex;
#pragma unroll
  for (int o = 16; o >= 1; o >>= 1) sm += __shfl_xor(sm, o, 32);
  out[(size_t)node * 32 + lane] = (v - mx) - __logf(sm);
}

// ---------------------------------------------------------------------------
extern "C" void kernel_launch(void* const* d_in, const int* in_sizes, int n_in,
                              void* d_out, int out_size, void* d_ws, size_t ws_size,
                              hipStream_t stream) {
  (void)n_in; (void)out_size; (void)ws_size;
  const float*     x   = (const float*)d_in[0];
  const long long* ei  = (const long long*)d_in[1];   // int64 [2, E]
  const float*     W1  = (const float*)d_in[2];
  const float*     a1s = (const float*)d_in[3];
  const float*     a1d = (const float*)d_in[4];
  const float*     W2  = (const float*)d_in[5];
  const float*     a2s = (const float*)d_in[6];
  const float*     a2d = (const float*)d_in[7];
  float* out = (float*)d_out;

  const int n = in_sizes[0] / 256;
  const int E = in_sizes[1] / 2;

  // workspace carve-up (~78 MB)
  float* ws = (float*)d_ws;
  float*    h1pre  = ws;            ws += (size_t)n * 64;
  float*    e1s    = ws;            ws += (size_t)n * 8;
  float*    e1d    = ws;            ws += (size_t)n * 8;
  unsigned* m1     = (unsigned*)ws; ws += (size_t)n * 8;
  float*    denom1 = ws;            ws += (size_t)n * 8;
  float*    h1     = ws;            ws += (size_t)n * 64;
  float*    h2pre  = ws;            ws += (size_t)n * 32;
  float*    e2s    = ws;            ws += (size_t)n;
  float*    e2d    = ws;            ws += (size_t)n;
  unsigned* m2     = (unsigned*)ws; ws += (size_t)n;
  float*    denom2 = ws;            ws += (size_t)n;

  const int tiles = (n + 15) / 16;
  const int gblk  = (tiles + 3) / 4;

  gat_init<<<(n * 64 + 255) / 256, 256, 0, stream>>>(denom1, m1, denom2, m2, h1, out, n);

  // layer 1
  gat_gemm1_wmma<<<gblk, 128, 0, stream>>>(x, W1, h1pre, n);
  gat_logits1<<<(n * 8 + 255) / 256, 256, 0, stream>>>(h1pre, a1s, a1d, e1s, e1d, n);
  gat_edge1_max    <<<(E * 8 + 255) / 256, 256, 0, stream>>>(ei, e1s, e1d, m1, E);
  gat_edge1_denom  <<<(E * 8 + 255) / 256, 256, 0, stream>>>(ei, e1s, e1d, m1, denom1, E);
  gat_edge1_scatter<<<(E * 8 + 255) / 256, 256, 0, stream>>>(ei, e1s, e1d, m1, denom1, h1pre, h1, E);
  gat_elu<<<(n * 64 + 255) / 256, 256, 0, stream>>>(h1, n);

  // layer 2
  gat_gemm2_wmma<<<gblk, 128, 0, stream>>>(h1, W2, h2pre, n);
  gat_logits2<<<(n + 255) / 256, 256, 0, stream>>>(h2pre, a2s, a2d, e2s, e2d, n);
  gat_edge2_max    <<<(E + 255) / 256, 256, 0, stream>>>(ei, e2s, e2d, m2, E);
  gat_edge2_denom  <<<(E + 255) / 256, 256, 0, stream>>>(ei, e2s, e2d, m2, denom2, E);
  gat_edge2_scatter<<<(E + 255) / 256, 256, 0, stream>>>(ei, e2s, e2d, m2, denom2, h2pre, out, E);

  gat_logsoftmax<<<(n + 3) / 4, 128, 0, stream>>>(out, n);
}
